// MemoryGate_44109314130761
// MI455X (gfx1250) — matrix operand order repaired and checked by reading, not compiled, not run on hardware
//
#include <hip/hip_runtime.h>
#include <hip/hip_bf16.h>
#include <stdint.h>
#include <stddef.h>

// ---------------- types ----------------
typedef __attribute__((ext_vector_type(16))) __bf16        v16bf;
typedef __attribute__((ext_vector_type(8)))  float         v8f;
typedef __attribute__((ext_vector_type(4)))  unsigned int  u32x4;

union FragB16 {           // 16 bf16 = 8 VGPRs (POD ext-vectors only)
  v16bf v;
  u32x4 q[2];
};

#define DIM    2048
#define KDIM   512
#define HALF   256
#define NKEYS  1024
#define NC     32
#define NROWS  8192   /* B*S */

__device__ __forceinline__ unsigned short f2bf(float f) {
  unsigned int u = __float_as_uint(f);
  u += 0x7FFFu + ((u >> 16) & 1u);          // round-to-nearest-even
  return (unsigned short)(u >> 16);
}

// ---------------- kernel A: fp32 -> bf16 for W and keys ----------------
__global__ void mg_convert_kernel(const float* __restrict__ W,
                                  const float* __restrict__ keys,
                                  unsigned short* __restrict__ Wb,
                                  unsigned short* __restrict__ keysb,
                                  int nW, int nK) {
  int i = blockIdx.x * blockDim.x + threadIdx.x;
  if (i < nW) Wb[i] = f2bf(W[i]);
  if (i < nK) keysb[i] = f2bf(keys[i]);
}

// ---------------- kernel B: Q = x * W^T  (bf16 WMMA, f32 acc) ----------------
// grid = 512 blocks (16 rows each), 256 threads = 8 waves, wave owns 4 N-tiles.
// X staging is double-buffered: chunk t+1's HBM loads issue before chunk t's
// WMMAs, conversion+LDS store happen after, one barrier per chunk.
__global__ void __launch_bounds__(256)
mg_gemm1_kernel(const float* __restrict__ x,
                const unsigned short* __restrict__ Wb,
                unsigned short* __restrict__ Qb) {
  const int tile = blockIdx.x;
  const int tid  = threadIdx.x;
  const int lane = tid & 31;
  const int wave = tid >> 5;
  const int hi   = lane >> 4;   // 0/1: which K-half this lane holds
  const int lm   = lane & 15;   // row (A) / column (B,C,D)

  __shared__ unsigned short Xs[2][16 * 128];   // double-buffered 16x128 bf16

  v8f acc[4];
  #pragma unroll
  for (int i = 0; i < 4; ++i) acc[i] = (v8f){0.f,0.f,0.f,0.f,0.f,0.f,0.f,0.f};

  const float* xrow = x + (size_t)tile * 16 * DIM;
  const int r = tid >> 4;          // 0..15 staging row
  const int c = (tid & 15) * 8;    // 0..120 staging col

  { // preload chunk 0
    const float* src = xrow + (size_t)r * DIM + c;
    unsigned short* dst = &Xs[0][r * 128 + c];
    #pragma unroll
    for (int j = 0; j < 8; ++j) dst[j] = f2bf(src[j]);
  }
  __syncthreads();

  for (int kci = 0; kci < 16; ++kci) {
    const int kc  = kci * 128;
    const int buf = kci & 1;

    // issue next chunk's global loads early (latency hidden under WMMAs)
    float nx[8];
    if (kci < 15) {
      const float* src = xrow + (size_t)r * DIM + (kc + 128) + c;
      #pragma unroll
      for (int j = 0; j < 8; ++j) nx[j] = src[j];
    }

    const unsigned short* Xb = &Xs[buf][0];
    #pragma unroll
    for (int ki = 0; ki < 4; ++ki) {
      const int kb = ki * 32;
      FragB16 a;   // A 16x32: lane row lm, K-chunks [hi*8,+8) and +16
      a.q[0] = *(const u32x4*)&Xb[lm * 128 + kb + hi * 8];
      a.q[1] = *(const u32x4*)&Xb[lm * 128 + kb + 16 + hi * 8];
      #pragma unroll
      for (int nti = 0; nti < 4; ++nti) {
        const int nt = wave * 4 + nti;
        const int n  = nt * 16 + lm;                  // output column / W row
        const int kg = kc + kb + hi * 8;              // K-chunks [hi*8,+8) and +16
        const unsigned short* wp = Wb + (size_t)n * DIM + kg;
        if (ki == 0) __builtin_prefetch(wp + 128, 0, 1);  // next K-chunk of W
        FragB16 b;
        b.q[0] = *(const u32x4*)(wp);
        b.q[1] = *(const u32x4*)(wp + 16);
        acc[nti] = __builtin_amdgcn_wmma_f32_16x16x32_bf16(
            false, a.v, false, b.v, (short)0, acc[nti], false, false);
      }
    }

    // convert + store next chunk into the other buffer
    if (kci < 15) {
      unsigned short* dst = &Xs[buf ^ 1][r * 128 + c];
      #pragma unroll
      for (int j = 0; j < 8; ++j) dst[j] = f2bf(nx[j]);
    }
    __syncthreads();
  }

  // D layout: c[j] -> (row j + 8*hi, col lm); store Q as bf16
  #pragma unroll
  for (int nti = 0; nti < 4; ++nti) {
    const int col = (wave * 4 + nti) * 16 + lm;
    #pragma unroll
    for (int j = 0; j < 8; ++j)
      Qb[(size_t)(tile * 16 + j + hi * 8) * KDIM + col] = f2bf(acc[nti][j]);
  }
}

// ---------------- per-row top-32 of 1024 via repeated wave-argmax ----------
__device__ __forceinline__ void mg_topk32(const float* __restrict__ sc_row,
                                          int lane,
                                          float* __restrict__ ts_out,
                                          int* __restrict__ ti_out) {
  float sv[32];
  #pragma unroll
  for (int j = 0; j < 32; ++j) sv[j] = sc_row[lane * 32 + j];
  unsigned int mask = 0;
  for (int it = 0; it < 32; ++it) {
    float v = -3.402823466e38f;
    int  bi = 0x7FFFFFFF;
    #pragma unroll
    for (int j = 0; j < 32; ++j)
      if (!((mask >> j) & 1u) && sv[j] > v) { v = sv[j]; bi = lane * 32 + j; }
    #pragma unroll
    for (int off = 16; off > 0; off >>= 1) {       // butterfly: all lanes get winner
      float ov = __shfl_xor(v, off, 32);
      int   oi = __shfl_xor(bi, off, 32);
      if (ov > v || (ov == v && oi < bi)) { v = ov; bi = oi; }  // stable ties
    }
    if ((bi >> 5) == lane) mask |= 1u << (bi & 31);
    if (lane == 0) { ts_out[it] = v; ti_out[it] = bi; }
  }
}

// ---------------- kernel C: scores GEMM + top-k + combine + softmax --------
__global__ void __launch_bounds__(256)
mg_gemm2_topk_kernel(const unsigned short* __restrict__ Qb,
                     const unsigned short* __restrict__ keysb,
                     int* __restrict__ out_idx,
                     float* __restrict__ out_scr) {
  const int tile = blockIdx.x;
  const int tid  = threadIdx.x;
  const int lane = tid & 31;
  const int wave = tid >> 5;
  const int hi   = lane >> 4;
  const int lm   = lane & 15;

  __shared__ unsigned short Qs[16 * KDIM];      // 16 KB
  __shared__ float          Sc[16 * NKEYS];     // 64 KB (reused for both halves)
  __shared__ float          TS[2][16 * NC];     // 4 KB
  __shared__ int            TI[2][16 * NC];     // 4 KB

  { // async-copy Q tile straight into LDS (no VGPR round trip, ASYNCcnt).
    // INST_OFFSET applies to both the LDS and global addresses (ISA 15.18.3),
    // so one base pair + offsets covers this thread's contiguous 64 bytes.
    const unsigned short* src = Qb + (size_t)tile * 16 * KDIM + tid * 32;
    unsigned int       lds = (unsigned int)(size_t)&Qs[tid * 32]; // low 32 bits = LDS offset
    unsigned long long ga  = (unsigned long long)(size_t)src;
    asm volatile(
        "global_load_async_to_lds_b128 %0, %1, off\n\t"
        "global_load_async_to_lds_b128 %0, %1, off offset:16\n\t"
        "global_load_async_to_lds_b128 %0, %1, off offset:32\n\t"
        "global_load_async_to_lds_b128 %0, %1, off offset:48"
        :: "v"(lds), "v"(ga) : "memory");
    asm volatile("s_wait_asynccnt 0x0" ::: "memory");
  }
  __syncthreads();

  for (int p = 0; p < 2; ++p) {
    // scores = Q_half (16x256) * keys_p^T (256x1024); 64 N-tiles, 8 per wave
    #pragma unroll
    for (int nti = 0; nti < 8; ++nti) {
      const int nt = wave * 8 + nti;
      v8f acc = (v8f){0.f,0.f,0.f,0.f,0.f,0.f,0.f,0.f};
      #pragma unroll
      for (int ks = 0; ks < 8; ++ks) {
        FragB16 a, b;
        const int col = p * HALF + ks * 32 + hi * 8;
        a.q[0] = *(const u32x4*)&Qs[lm * KDIM + col];
        a.q[1] = *(const u32x4*)&Qs[lm * KDIM + col + 16];
        const int n  = nt * 16 + lm;
        const int kg = ks * 32 + hi * 8;
        const unsigned short* kp =
            keysb + (size_t)p * NKEYS * HALF + (size_t)n * HALF + kg;
        b.q[0] = *(const u32x4*)(kp);
        b.q[1] = *(const u32x4*)(kp + 16);
        acc = __builtin_amdgcn_wmma_f32_16x16x32_bf16(
            false, a.v, false, b.v, (short)0, acc, false, false);
      }
      #pragma unroll
      for (int j = 0; j < 8; ++j)
        Sc[(j + hi * 8) * NKEYS + nt * 16 + lm] = acc[j];
    }
    __syncthreads();

    #pragma unroll
    for (int r2 = 0; r2 < 2; ++r2) {       // 8 waves x 2 rows = 16 rows
      const int row = wave * 2 + r2;
      mg_topk32(&Sc[row * NKEYS], lane, &TS[p][row * NC], &TI[p][row * NC]);
    }
    __syncthreads();    // also protects Sc before next half overwrites it
  }

  // combined 32x32 outer-sum top-32 + softmax; lane L owns i=L, j in [0,32)
  for (int r2 = 0; r2 < 2; ++r2) {
    const int row = wave * 2 + r2;
    const float t1 = TS[0][row * NC + lane];
    const float t2 = TS[1][row * NC + lane];
    const int   i1 = TI[0][row * NC + lane];
    const int   i2 = TI[1][row * NC + lane];
    unsigned int mask = 0;
    float sel_v = 0.0f;
    int   sel_i = 0;
    for (int it = 0; it < NC; ++it) {
      float v = -3.402823466e38f;
      int  bi = 0x7FFFFFFF;
      #pragma unroll
      for (int j = 0; j < 32; ++j) {
        const float cv = t1 + __shfl(t2, j, 32);   // uniform shuffle, all lanes
        if (!((mask >> j) & 1u) && cv > v) { v = cv; bi = lane * 32 + j; }
      }
      #pragma unroll
      for (int off = 16; off > 0; off >>= 1) {
        float ov = __shfl_xor(v, off, 32);
        int   oi = __shfl_xor(bi, off, 32);
        if (ov > v || (ov == v && oi < bi)) { v = ov; bi = oi; }
      }
      if ((bi >> 5) == lane) mask |= 1u << (bi & 31);
      const int ki1 = __shfl(i1, bi >> 5, 32);   // all lanes active for shfl
      const int ki2 = __shfl(i2, bi & 31, 32);
      if (lane == it) { sel_v = v; sel_i = ki1 * NKEYS + ki2; }
    }
    // softmax over the 32 selected (descending, so max lives in lane 0)
    const float m = __shfl(sel_v, 0, 32);
    const float e = __expf(sel_v - m);
    float s = e;
    #pragma unroll
    for (int off = 16; off > 0; off >>= 1) s += __shfl_xor(s, off, 32);
    const size_t o = (size_t)(tile * 16 + row) * NC + lane;
    out_idx[o] = sel_i;
    out_scr[o] = e / s;
  }
}

// ---------------- launch ----------------
extern "C" void kernel_launch(void* const* d_in, const int* in_sizes, int n_in,
                              void* d_out, int out_size, void* d_ws, size_t ws_size,
                              hipStream_t stream) {
  const float* x    = (const float*)d_in[0];   // (4,2048,2048)
  const float* W    = (const float*)d_in[1];   // (512,2048)
  const float* keys = (const float*)d_in[2];   // (2,1024,256)

  // workspace layout (bf16 copies + Q), ~11 MiB total
  unsigned short* Wb    = (unsigned short*)d_ws;            // 512*2048
  unsigned short* keysb = Wb + (size_t)KDIM * DIM;          // 2*1024*256
  unsigned short* Qb    = keysb + (size_t)2 * NKEYS * HALF; // 8192*512

  int*   out_idx = (int*)d_out;                             // B*S*32 int32
  float* out_scr = (float*)d_out + (size_t)NROWS * NC;      // B*S*32 f32

  const int nW = KDIM * DIM;            // 1,048,576
  const int nK = 2 * NKEYS * HALF;      //   524,288
  mg_convert_kernel<<<(nW + 255) / 256, 256, 0, stream>>>(W, keys, Wb, keysb, nW, nK);
  mg_gemm1_kernel<<<NROWS / 16, 256, 0, stream>>>(x, Wb, Qb);
  mg_gemm2_topk_kernel<<<NROWS / 16, 256, 0, stream>>>(Qb, keysb, out_idx, out_scr);
}
